// ScalableDAGv3_23261542875309
// MI455X (gfx1250) — compile-verified, matrix operand order checked by reading, not run on hardware
//
#include <hip/hip_runtime.h>
#include <hip/hip_bf16.h>
#include <math.h>

typedef __attribute__((ext_vector_type(16))) _Float16 v16h;
typedef __attribute__((ext_vector_type(8)))  float    v8f;

#define B_SZ  256
#define D_SZ  128
#define K_SZ  32
#define M1_SZ 64
#define M2_SZ 32
#define H1_SZ 128
#define H2_SZ 64
#define L_SZ  32

union H2u { unsigned u; _Float16 h[2]; };

// Assemble a 16x32 f16 WMMA operand fragment (A, or column-major-staged B).
// Memory layout: row-major [rows][row_stride] f16, fragment rows selected by
// caller-computed per-lane 'row', K columns at col_base..col_base+31.
// Per ISA 7.12.2 (16-bit A 16x32): VGPR v holds a K pair; lanes 16-31 offset K by 8.
__device__ __forceinline__ v16h load_frag(const _Float16* __restrict__ p,
                                          int row, int row_stride,
                                          int col_base, int lane) {
  const int half = lane >> 4;
  v16h f;
#pragma unroll
  for (int v = 0; v < 8; ++v) {
    const int k0 = ((v >> 2) << 4) + ((v & 3) << 1) + (half << 3);
    H2u t;
    t.u = *reinterpret_cast<const unsigned*>(p + row * row_stride + col_base + k0);
    f[2 * v]     = t.h[0];
    f[2 * v + 1] = t.h[1];
  }
  return f;
}

__device__ __forceinline__ float fast_sigmoid(float x) {
  return 1.0f / (1.0f + __expf(-x));
}

// ---------------- prep kernels ----------------

// W1 = W1_pos - W1_neg  -> f16 [jh=8192][i=128]; extract diag column + fused bias
__global__ __launch_bounds__(256) void k_prep_w1(
    const float* __restrict__ wp, const float* __restrict__ wn,
    const float* __restrict__ bp, const float* __restrict__ bn,
    _Float16* __restrict__ w1h, float* __restrict__ diagW, float* __restrict__ b1) {
  const int idx = blockIdx.x * blockDim.x + threadIdx.x;   // < 128*64*128
  const int i  = idx & 127;
  const int jh = idx >> 7;         // j*64 + h
  const int j  = jh >> 6;
  const float w = wp[idx] - wn[idx];
  w1h[idx] = (_Float16)w;
  if (i == j) diagW[jh] = w;
  if (i == 0) b1[jh] = bp[jh] - bn[jh];
}

__global__ __launch_bounds__(256) void k_prep_x(
    const float* __restrict__ x, _Float16* __restrict__ xh) {
  const int idx = blockIdx.x * blockDim.x + threadIdx.x;   // < 32768
  xh[idx] = (_Float16)x[idx];
}

// lc1_w [k][m][n] -> K-contiguous f16 [col=k*32+n][m=64]
__global__ __launch_bounds__(256) void k_prep_l1(
    const float* __restrict__ lc1, _Float16* __restrict__ l1t) {
  const int idx = blockIdx.x * blockDim.x + threadIdx.x;   // < 32*64*32
  const int k = idx >> 11;
  const int m = (idx >> 5) & 63;
  const int n = idx & 31;
  l1t[(k * 32 + n) * 64 + m] = (_Float16)lc1[idx];
}

// ---------------- GEMM1: x @ W1^T, fused diag-fix + bias + sigmoid -> S (f16) ----------------
// C[b, jh] = sum_i x[b,i]*W1[jh,i]; M=256, N=8192, K=128.
// One wave per (16-row tile, 64-column group): A fragment reused across 4 N tiles,
// 16 WMMAs per wave. The 64-col group is j-aligned so the diag column is wave-constant.
__global__ __launch_bounds__(256) void k_gemm1(
    const _Float16* __restrict__ xh, const _Float16* __restrict__ w1h,
    const float* __restrict__ xf, const float* __restrict__ diagW,
    const float* __restrict__ b1, _Float16* __restrict__ S) {
  const int wid  = (blockIdx.x * blockDim.x + threadIdx.x) >> 5;
  const int lane = threadIdx.x & 31;
  const int mt = wid & 15;     // 16 row tiles over B=256
  const int ng = wid >> 4;     // 128 col groups of 64 over 8192
  const int half = lane >> 4;
  const int l15  = lane & 15;

  const int row_a = mt * 16 + l15;   // batch row

  v8f c[4] = {};
#pragma unroll
  for (int kk = 0; kk < 4; ++kk) {
    const v16h a = load_frag(xh, row_a, D_SZ, kk * 32, lane);
#pragma unroll
    for (int n = 0; n < 4; ++n) {
      const v16h b = load_frag(w1h, ng * 64 + n * 16 + l15, D_SZ, kk * 32, lane);
      c[n] = __builtin_amdgcn_wmma_f32_16x16x32_f16(false, a, false, b, (short)0,
                                                    c[n], false, false);
    }
  }

  // epilogue: phi = c - x[b,j]*diagW[jh] + b1[jh]; S = f16(sigmoid(phi))
  const int j = ng;                         // jh>>6 is constant over the 64-col group
  float xv[8];
#pragma unroll
  for (int v = 0; v < 8; ++v)
    xv[v] = xf[(mt * 16 + v + 8 * half) * D_SZ + j];

#pragma unroll
  for (int n = 0; n < 4; ++n) {
    const int jh = ng * 64 + n * 16 + l15;
    const float dw = diagW[jh];
    const float bb = b1[jh];
#pragma unroll
    for (int v = 0; v < 8; ++v) {
      const int brow = mt * 16 + v + 8 * half;            // C/D layout: row = v + 8*half
      const float phi = c[n][v] - xv[v] * dw + bb;
      S[brow * (D_SZ * M1_SZ) + jh] = (_Float16)fast_sigmoid(phi);
    }
  }
}

// ---------------- GEMM2: S[32768,64] @ L1t -> fused bias+sigmoid+LC2 dot -> last_phi ----------------
// One wave per (16-row tile, group of 4 k-copies): A loaded once per K-step and
// reused across 8 B fragments -> 16 WMMAs per wave. Each k-copy's M2=32 columns
// live in one wave, so the LC2 reduction is an in-wave shfl_xor butterfly.
__global__ __launch_bounds__(256) void k_gemm2(
    const _Float16* __restrict__ S, const _Float16* __restrict__ l1t,
    const float* __restrict__ lc1_b, const float* __restrict__ lc2_w,
    const float* __restrict__ lc2_b, float* __restrict__ lastphi) {
  const int wid  = (blockIdx.x * blockDim.x + threadIdx.x) >> 5;
  const int lane = threadIdx.x & 31;
  const int rt = wid >> 3;     // 2048 row tiles over B*D=32768
  const int kg = wid & 7;      // 8 groups of 4 k-copies
  const int half = lane >> 4;
  const int l15  = lane & 15;

  const int row_a = rt * 16 + l15;

  v8f c[4][2] = {};
#pragma unroll
  for (int kk = 0; kk < 2; ++kk) {
    const v16h a = load_frag(S, row_a, M1_SZ, kk * 32, lane);
#pragma unroll
    for (int q = 0; q < 4; ++q) {
      const int kk2 = kg * 4 + q;
      const v16h b0 = load_frag(l1t, kk2 * 32 + l15,      M1_SZ, kk * 32, lane);
      const v16h b1f = load_frag(l1t, kk2 * 32 + 16 + l15, M1_SZ, kk * 32, lane);
      c[q][0] = __builtin_amdgcn_wmma_f32_16x16x32_f16(false, a, false, b0, (short)0,
                                                       c[q][0], false, false);
      c[q][1] = __builtin_amdgcn_wmma_f32_16x16x32_f16(false, a, false, b1f, (short)0,
                                                       c[q][1], false, false);
    }
  }

  // epilogue: sig(c + lc1_b) * lc2_w, reduce over n (lane dim of C/D), + lc2_b
#pragma unroll
  for (int q = 0; q < 4; ++q) {
    const int kk2 = kg * 4 + q;
    const int n0 = kk2 * 32 + l15;
    const int n1 = n0 + 16;
    const float bias0 = lc1_b[n0], bias1 = lc1_b[n1];
    const float w0 = lc2_w[n0],    w1 = lc2_w[n1];

    float t[8];
#pragma unroll
    for (int v = 0; v < 8; ++v) {
      t[v] = fast_sigmoid(c[q][0][v] + bias0) * w0 +
             fast_sigmoid(c[q][1][v] + bias1) * w1;
    }
#pragma unroll
    for (int off = 1; off < 16; off <<= 1) {
#pragma unroll
      for (int v = 0; v < 8; ++v) t[v] += __shfl_xor(t[v], off, 32);
    }
    if (l15 == 0) {
      const float b2 = lc2_b[kk2];
#pragma unroll
      for (int v = 0; v < 8; ++v) {
        const int bd = rt * 16 + v + 8 * half;
        lastphi[bd * K_SZ + kk2] = t[v] + b2;
      }
    }
  }
}

// ---------------- VAE: tiny MLP chain, one block per batch row ----------------
__global__ __launch_bounds__(128) void k_vae(
    const float* __restrict__ beta_w,
    const float* __restrict__ enc1_w, const float* __restrict__ enc1_b,
    const float* __restrict__ enc2_w, const float* __restrict__ enc2_b,
    const float* __restrict__ mu_w,  const float* __restrict__ mu_b,
    const float* __restrict__ lv_w,  const float* __restrict__ lv_b,
    const float* __restrict__ dec1_w, const float* __restrict__ dec1_b,
    const float* __restrict__ dec2_w, const float* __restrict__ dec2_b,
    const float* __restrict__ dec3_w, const float* __restrict__ dec3_b,
    const float* __restrict__ eps,
    float* __restrict__ out_mu, float* __restrict__ out_lv,
    float* __restrict__ recon) {
  __shared__ float sA[H1_SZ];
  __shared__ float sB[H1_SZ];
  const int b = blockIdx.x, t = threadIdx.x;

  if (t < K_SZ) sA[t] = beta_w[b * K_SZ + t];
  __syncthreads();
  {  // enc1: 32 -> 128, relu
    float acc = enc1_b[t];
    for (int i = 0; i < K_SZ; ++i) acc += sA[i] * enc1_w[i * H1_SZ + t];
    sB[t] = fmaxf(acc, 0.0f);
  }
  __syncthreads();
  if (t < H2_SZ) {  // enc2: 128 -> 64, relu
    float acc = enc2_b[t];
    for (int i = 0; i < H1_SZ; ++i) acc += sB[i] * enc2_w[i * H2_SZ + t];
    sA[t] = fmaxf(acc, 0.0f);
  }
  __syncthreads();
  if (t < L_SZ) {  // mu / logvar / reparam z
    float mu = mu_b[t], lv = lv_b[t];
    for (int i = 0; i < H2_SZ; ++i) {
      mu += sA[i] * mu_w[i * L_SZ + t];
      lv += sA[i] * lv_w[i * L_SZ + t];
    }
    out_mu[b * L_SZ + t] = mu;
    out_lv[b * L_SZ + t] = lv;
    sB[t] = mu + eps[b * L_SZ + t] * expf(0.5f * lv);
  }
  __syncthreads();
  if (t < H2_SZ) {  // dec1: 32 -> 64, relu
    float acc = dec1_b[t];
    for (int i = 0; i < L_SZ; ++i) acc += sB[i] * dec1_w[i * H2_SZ + t];
    sA[t] = fmaxf(acc, 0.0f);
  }
  __syncthreads();
  {  // dec2: 64 -> 128, relu
    float acc = dec2_b[t];
    for (int i = 0; i < H2_SZ; ++i) acc += sA[i] * dec2_w[i * H1_SZ + t];
    sB[t] = fmaxf(acc, 0.0f);
  }
  __syncthreads();
  if (t < K_SZ) {  // dec3: 128 -> 32
    float acc = dec3_b[t];
    for (int i = 0; i < H1_SZ; ++i) acc += sB[i] * dec3_w[i * K_SZ + t];
    recon[b * K_SZ + t] = acc;
  }
}

// ---------------- final projections: y1, y2 ----------------
__global__ __launch_bounds__(256) void k_final(
    const float* __restrict__ lastphi, const float* __restrict__ beta_w,
    const float* __restrict__ beta_b, const float* __restrict__ recon,
    float* __restrict__ y1, float* __restrict__ y2) {
  const int idx = blockIdx.x * blockDim.x + threadIdx.x;  // b*128 + d
  const int b = idx >> 7;
  const float* lp = lastphi + idx * K_SZ;
  const float* bw = beta_w + b * K_SZ;
  const float* rc = recon + b * K_SZ;
  float a1 = 0.0f, a2 = 0.0f;
#pragma unroll
  for (int k = 0; k < K_SZ; ++k) {
    const float v = lp[k];
    a1 += v * bw[k];
    a2 += v * rc[k];
  }
  const float bb = beta_b[b];
  y1[idx] = a1 + bb;
  y2[idx] = a2 + bb;
}

extern "C" void kernel_launch(void* const* d_in, const int* in_sizes, int n_in,
                              void* d_out, int out_size, void* d_ws, size_t ws_size,
                              hipStream_t stream) {
  const float* x      = (const float*)d_in[0];
  const float* W1p    = (const float*)d_in[1];
  const float* W1n    = (const float*)d_in[2];
  const float* b1p    = (const float*)d_in[3];
  const float* b1n    = (const float*)d_in[4];
  const float* lc1_w  = (const float*)d_in[5];
  const float* lc1_b  = (const float*)d_in[6];
  const float* lc2_w  = (const float*)d_in[7];
  const float* lc2_b  = (const float*)d_in[8];
  const float* beta_w = (const float*)d_in[9];
  const float* beta_b = (const float*)d_in[10];
  const float* enc1_w = (const float*)d_in[11];
  const float* enc1_b = (const float*)d_in[12];
  const float* enc2_w = (const float*)d_in[13];
  const float* enc2_b = (const float*)d_in[14];
  const float* mu_w   = (const float*)d_in[15];
  const float* mu_b   = (const float*)d_in[16];
  const float* lv_w   = (const float*)d_in[17];
  const float* lv_b   = (const float*)d_in[18];
  const float* dec1_w = (const float*)d_in[19];
  const float* dec1_b = (const float*)d_in[20];
  const float* dec2_w = (const float*)d_in[21];
  const float* dec2_b = (const float*)d_in[22];
  const float* dec3_w = (const float*)d_in[23];
  const float* dec3_b = (const float*)d_in[24];
  const float* eps    = (const float*)d_in[25];

  // workspace carve-up (all offsets 256B-aligned)
  char* ws = (char*)d_ws;
  _Float16* W1h     = (_Float16*)(ws + 0);                  // 8192*128 f16 = 2 MB
  _Float16* xh      = (_Float16*)(ws + 2097152);            // 32768 f16
  float*    diagW   = (float*)   (ws + 2162688);            // 8192 f32
  float*    b1f     = (float*)   (ws + 2195456);            // 8192 f32
  _Float16* S       = (_Float16*)(ws + 2228224);            // 32768*64 f16 = 4 MB
  _Float16* L1t     = (_Float16*)(ws + 6422528);            // 1024*64 f16
  float*    lastphi = (float*)   (ws + 6553600);            // 32768*32 f32 = 4 MB
  float*    recon   = (float*)   (ws + 10747904);           // 256*32 f32

  float* y1     = (float*)d_out;            // [256,128]
  float* y2     = (float*)d_out + 32768;    // [256,128]
  float* out_mu = (float*)d_out + 65536;    // [256,32]
  float* out_lv = (float*)d_out + 73728;    // [256,32]

  k_prep_w1<<<4096, 256, 0, stream>>>(W1p, W1n, b1p, b1n, W1h, diagW, b1f);
  k_prep_x <<<128, 256, 0, stream>>>(x, xh);
  k_prep_l1<<<256, 256, 0, stream>>>(lc1_w, L1t);

  // GEMM1: 16 row tiles x 128 col groups = 2048 waves, 8 waves/block
  k_gemm1<<<256, 256, 0, stream>>>(xh, W1h, x, diagW, b1f, S);

  // GEMM2: 2048 row tiles x 8 k-groups = 16384 waves, 8 waves/block
  k_gemm2<<<2048, 256, 0, stream>>>(S, L1t, lc1_b, lc2_w, lc2_b, lastphi);

  k_vae<<<256, 128, 0, stream>>>(beta_w, enc1_w, enc1_b, enc2_w, enc2_b,
                                 mu_w, mu_b, lv_w, lv_b, dec1_w, dec1_b,
                                 dec2_w, dec2_b, dec3_w, dec3_b, eps,
                                 out_mu, out_lv, recon);

  k_final<<<128, 256, 0, stream>>>(lastphi, beta_w, beta_b, recon, y1, y2);
}
